// GraphConvPoolNN_83854941487716
// MI455X (gfx1250) — matrix operand order repaired and checked by reading, not compiled, not run on hardware
//
#include <hip/hip_runtime.h>
#include <hip/hip_bf16.h>
#include <math.h>

// Problem constants (match reference)
#define NN   65536
#define EE   262144
#define FINF 32
#define HIDN 64
#define CATF 96     // FINF + HIDN
#define CAT2 160    // CATF + HIDN

#define CDIV(a,b) (((a)+(b)-1)/(b))

typedef __attribute__((ext_vector_type(16))) _Float16 v16h;
typedef __attribute__((ext_vector_type(8)))  float    v8f;
typedef unsigned long long u64;

// ---------------------------------------------------------------------------
// Utility kernels
// ---------------------------------------------------------------------------
__global__ void k_fill_f32(float* p, float v, int n){
  int i = blockIdx.x*blockDim.x + threadIdx.x;
  if(i < n) p[i] = v;
}
__global__ void k_fill_i32(int* p, int v, int n){
  int i = blockIdx.x*blockDim.x + threadIdx.x;
  if(i < n) p[i] = v;
}
__global__ void k_cvt_f16(const float* __restrict__ in, _Float16* __restrict__ out, int n){
  int i = blockIdx.x*blockDim.x + threadIdx.x;
  if(i < n) out[i] = (_Float16)in[i];
}
__global__ void k_split_edges(const int* __restrict__ ed, int* s, int* d, float* w){
  int e = blockIdx.x*blockDim.x + threadIdx.x;
  if(e < EE){ s[e] = ed[2*e]; d[e] = ed[2*e+1]; w[e] = 1.f; }
}
// out[i, 0:fa] = A[i,:], out[i, fa:fa+fb] = B[i,:]
__global__ void k_concat2(const float* __restrict__ A, int fa,
                          const float* __restrict__ B, int fb,
                          float* __restrict__ out, int n){
  int ftot = fa + fb;
  int idx = blockIdx.x*blockDim.x + threadIdx.x;
  if(idx < n*ftot){
    int i = idx / ftot, j = idx % ftot;
    out[idx] = (j < fa) ? A[i*fa + j] : B[i*fb + (j - fa)];
  }
}

// ---------------------------------------------------------------------------
// WMMA dense GEMM: C[NN x 64] = A[NN x K] * B[K x 64]  (+bias)(+relu)
// A: f32 row-major, B: f16 row-major (preconverted), C: f32. K % 32 == 0.
// Block = 128 threads (4 waves). Block owns a 16-row stripe; wave w owns
// the 16-col tile [16w, 16w+16).
//
// Tiles are staged into LDS in *WMMA fragment order* so each lane's A/B
// fragment is one contiguous 32-byte chunk (-> ds_load_b128 pairs instead
// of 16 scalar ds_load_u16).
//   A fragment (16-bit A 16x32): element (m,k) -> lane m + 16*(k>>4),
//                                elem k&15
//   B fragment (16-bit B 32x16): element (k,n) -> lane (n&15) + 16*(k>>4),
//                                elem k&15, per 16-col wave tile.
// ---------------------------------------------------------------------------
template<int K>
__global__ void k_wmma_mm(const float* __restrict__ A, const _Float16* __restrict__ B,
                          const float* __restrict__ bias, float* __restrict__ C,
                          int relu){
  static_assert(K % 32 == 0, "K must be a multiple of 32");
  __shared__ alignas(16) _Float16 Af[32][16];      // fragment-ordered A tile
  __shared__ alignas(16) _Float16 Bf[4][32][16];   // fragment-ordered B tiles
  const int tid  = threadIdx.x;
  const int wave = tid >> 5;        // 0..3 -> column tile
  const int lane = tid & 31;
  const int row0 = blockIdx.x * 16;

  // Prefetch the next block's A stripe into cache (global_prefetch_b8).
  if(row0 + 16 < NN)
    __builtin_prefetch(A + (size_t)(row0 + 16)*K, 0, 3);

  v8f acc = {};
  #pragma unroll
  for(int k0 = 0; k0 < K; k0 += 32){
    // --- stage A tile (16 rows x 32 K), vector float4 loads, 4 elems/thread
    {
      int linear = tid << 2;                    // 0..508
      int r = linear >> 5;                      // row in tile
      int c = linear & 31;                      // k offset base
      float4 av = *(const float4*)(A + (size_t)(row0 + r)*K + k0 + c);
      float vals[4] = {av.x, av.y, av.z, av.w};
      #pragma unroll
      for(int j = 0; j < 4; ++j){
        int k = c + j;
        Af[r + ((k >> 4) << 4)][k & 15] = (_Float16)vals[j];
      }
    }
    // --- stage B tile (32 K x 64 cols), 16 halves (32B) per thread
    {
      int k  = tid >> 2;                        // 0..31
      int cg = (tid & 3) << 4;                  // column group base 0/16/32/48
      union { uint4 u[2]; _Float16 h[16]; } tmp;
      const uint4* src = (const uint4*)(B + (size_t)(k0 + k)*64 + cg);
      tmp.u[0] = src[0];
      tmp.u[1] = src[1];
      int wt   = cg >> 4;
      int lrow = (k >> 4) << 4;                 // lane offset 0 or 16
      int e    = k & 15;
      #pragma unroll
      for(int n = 0; n < 16; ++n)
        Bf[wt][n + lrow][e] = tmp.h[n];
    }
    __syncthreads();

    v16h af = *(const v16h*)(&Af[lane][0]);
    v16h bf = *(const v16h*)(&Bf[wave][lane][0]);
    acc = __builtin_amdgcn_wmma_f32_16x16x32_f16(false, af, false, bf,
                                                 (short)0, acc, false, false);
    __syncthreads();
  }
  // C layout: lanes 0-15 -> M 0..7 (vgpr r), lanes 16-31 -> M 8..15; N = lane&15.
  {
    int n  = (wave << 4) + (lane & 15);
    int mo = (lane >> 4) << 3;
    float b = bias ? bias[n] : 0.f;
    #pragma unroll
    for(int r = 0; r < 8; ++r){
      float v = acc[r] + b;
      if(relu) v = fmaxf(v, 0.f);
      C[(size_t)(row0 + mo + r)*64 + n] = v;
    }
  }
}

// ---------------------------------------------------------------------------
// GCN aggregation (symmetric normalization with forced self loops)
// ---------------------------------------------------------------------------
__global__ void k_deg(const int* __restrict__ s, const int* __restrict__ d,
                      const float* __restrict__ w, float* __restrict__ deg){
  int e = blockIdx.x*blockDim.x + threadIdx.x;
  if(e < EE){
    int a = s[e], b = d[e]; float we = w[e];
    if(we > 0.f && a != b) atomicAdd(&deg[b], we);
  }
}
__global__ void k_dis(const float* __restrict__ deg, float* __restrict__ dis){
  int i = blockIdx.x*blockDim.x + threadIdx.x;
  if(i < NN) dis[i] = rsqrtf(deg[i]);           // deg >= 1 always (self loop)
}
__global__ void k_self_init(const float* __restrict__ h, const float* __restrict__ dis,
                            float* __restrict__ out){
  int idx = blockIdx.x*blockDim.x + threadIdx.x;
  if(idx < NN*HIDN){
    int i = idx >> 6;
    out[idx] = h[idx] * dis[i] * dis[i];
  }
}
__global__ void k_edge_agg(const float* __restrict__ h, const int* __restrict__ s,
                           const int* __restrict__ d, const float* __restrict__ w,
                           const float* __restrict__ dis, float* __restrict__ out){
  int e = blockIdx.x*blockDim.x + threadIdx.x;
  if(e < EE){
    int a = s[e], b = d[e]; float we = w[e];
    if(we > 0.f && a != b){
      float c = dis[a] * we * dis[b];
      #pragma unroll 4
      for(int f = 0; f < HIDN; ++f)
        atomicAdd(&out[(size_t)b*HIDN + f], h[(size_t)a*HIDN + f] * c);
    }
  }
}
__global__ void k_bias_relu(float* __restrict__ out, const float* __restrict__ bias){
  int idx = blockIdx.x*blockDim.x + threadIdx.x;
  if(idx < NN*HIDN){
    int f = idx & 63;
    out[idx] = fmaxf(out[idx] + bias[f], 0.f);
  }
}

// ---------------------------------------------------------------------------
// EdgePool kernels
// ---------------------------------------------------------------------------
__device__ __forceinline__ void atomicMaxFloat(float* addr, float v){
  if(v >= 0.f) atomicMax((int*)addr, __float_as_int(v));
  else         atomicMin((unsigned int*)addr, (unsigned int)__float_as_int(v));
}

__global__ void k_pool_raw(const float* __restrict__ x, const int* __restrict__ s,
                           const int* __restrict__ d, const float* __restrict__ w,
                           const float* __restrict__ pw, const float* __restrict__ pb,
                           float* __restrict__ r){
  int e = blockIdx.x*blockDim.x + threadIdx.x;
  if(e < EE){
    int a = s[e], b = d[e];
    float acc = pb[0];
    const float* xs = x + (size_t)a*CATF;
    const float* xd = x + (size_t)b*CATF;
    #pragma unroll 4
    for(int j = 0; j < CATF; ++j) acc += xs[j]*pw[j];
    #pragma unroll 4
    for(int j = 0; j < CATF; ++j) acc += xd[j]*pw[CATF + j];
    r[e] = (w[e] > 0.f) ? acc : -1e30f;
  }
}
__global__ void k_segmax(const float* __restrict__ r, const int* __restrict__ d,
                         float* __restrict__ m){
  int e = blockIdx.x*blockDim.x + threadIdx.x;
  if(e < EE) atomicMaxFloat(&m[d[e]], r[e]);
}
__global__ void k_segden(const float* __restrict__ r, const int* __restrict__ d,
                         const float* __restrict__ w, const float* __restrict__ m,
                         float* __restrict__ den){
  int e = blockIdx.x*blockDim.x + threadIdx.x;
  if(e < EE && w[e] > 0.f) atomicAdd(&den[d[e]], expf(r[e] - m[d[e]]));
}
__global__ void k_score(const float* __restrict__ r, const int* __restrict__ d,
                        const float* __restrict__ w, const float* __restrict__ m,
                        const float* __restrict__ den, float* __restrict__ score){
  int e = blockIdx.x*blockDim.x + threadIdx.x;
  if(e < EE){
    if(w[e] > 0.f){
      int b = d[e];
      score[e] = expf(r[e] - m[b]) / fmaxf(den[b], 1e-16f) + 0.5f;
    } else score[e] = -__builtin_inff();
  }
}
// keys = ordered_uint(-score) << 32 | e   (ascending u64 sort == stable argsort(-score))
__global__ void k_sortkey_score(const float* __restrict__ score, u64* __restrict__ keys){
  int e = blockIdx.x*blockDim.x + threadIdx.x;
  if(e < EE){
    unsigned ub = __float_as_uint(-score[e]);
    ub = (ub & 0x80000000u) ? ~ub : (ub | 0x80000000u);
    keys[e] = ((u64)ub << 32) | (unsigned)e;
  }
}
__global__ void k_bitonic(u64* __restrict__ keys, unsigned j, unsigned k){
  unsigned i = blockIdx.x*blockDim.x + threadIdx.x;
  unsigned ixj = i ^ j;
  if(ixj > i){
    u64 a = keys[i], b = keys[ixj];
    bool up = ((i & k) == 0);
    if((a > b) == up){ keys[i] = b; keys[ixj] = a; }
  }
}
__global__ void k_gather_sorted(const u64* __restrict__ keys, const int* __restrict__ s,
                                const int* __restrict__ d, const float* __restrict__ w,
                                const float* __restrict__ score,
                                int* __restrict__ ssrt, int* __restrict__ dsrt,
                                int* __restrict__ vsrt, float* __restrict__ scsrt){
  int i = blockIdx.x*blockDim.x + threadIdx.x;
  if(i < EE){
    int idx = (int)(keys[i] & 0xFFFFFFFFu);
    ssrt[i]  = s[idx];
    dsrt[i]  = d[idx];
    vsrt[i]  = (w[idx] > 0.f) ? 1 : 0;
    scsrt[i] = score[idx];
  }
}
// Strictly sequential greedy matching (lax.scan equivalent).
__global__ void k_greedy(const int* __restrict__ ssrt, const int* __restrict__ dsrt,
                         const int* __restrict__ vsrt, int* __restrict__ assigned,
                         int* __restrict__ ok, int* __restrict__ cid, int* __restrict__ kOut){
  if(blockIdx.x != 0 || threadIdx.x != 0) return;
  int k = 0;
  for(int e = 0; e < EE; ++e){
    int a = ssrt[e], b = dsrt[e];
    int o = vsrt[e] && !assigned[a] && !assigned[b];
    ok[e]  = o;
    cid[e] = k;                       // count BEFORE increment, like the scan carry
    if(o){ assigned[a] = 1; assigned[b] = 1; ++k; }
  }
  *kOut = k;
}
__global__ void k_scatter_cluster(const int* __restrict__ ssrt, const int* __restrict__ dsrt,
                                  const int* __restrict__ ok, const int* __restrict__ cid,
                                  const float* __restrict__ scsrt,
                                  int* __restrict__ cluster, float* __restrict__ csc){
  int i = blockIdx.x*blockDim.x + threadIdx.x;
  if(i < EE && ok[i]){
    int c = cid[i];
    cluster[ssrt[i]] = c;
    cluster[dsrt[i]] = c;
    csc[c] = scsrt[i];
  }
}
// 3-kernel inclusive scan of (!assigned) over NN (256 blocks of 256)
__global__ void k_scan_local(const int* __restrict__ assigned, int* __restrict__ partial,
                             int* __restrict__ bsums){
  __shared__ int sh[256];
  int i = blockIdx.x*256 + threadIdx.x;
  int v = assigned[i] ? 0 : 1;
  sh[threadIdx.x] = v; __syncthreads();
  for(int off = 1; off < 256; off <<= 1){
    int t = (threadIdx.x >= off) ? sh[threadIdx.x - off] : 0;
    __syncthreads();
    sh[threadIdx.x] += t;
    __syncthreads();
  }
  partial[i] = sh[threadIdx.x];
  if(threadIdx.x == 255) bsums[blockIdx.x] = sh[255];
}
__global__ void k_scan_block(int* __restrict__ bsums){
  __shared__ int sh[256];
  int v = bsums[threadIdx.x];
  sh[threadIdx.x] = v; __syncthreads();
  for(int off = 1; off < 256; off <<= 1){
    int t = (threadIdx.x >= off) ? sh[threadIdx.x - off] : 0;
    __syncthreads();
    sh[threadIdx.x] += t;
    __syncthreads();
  }
  bsums[threadIdx.x] = sh[threadIdx.x] - v;   // exclusive
}
__global__ void k_final_cluster(const int* __restrict__ assigned, const int* __restrict__ partial,
                                const int* __restrict__ bsums, const int* __restrict__ kdev,
                                int* __restrict__ cluster){
  int i = blockIdx.x*blockDim.x + threadIdx.x;
  if(i < NN && !assigned[i])
    cluster[i] = kdev[0] + bsums[i >> 8] + partial[i] - 1;
}
__global__ void k_newx_sum(const float* __restrict__ x, const int* __restrict__ cluster,
                           float* __restrict__ newx){
  int i = blockIdx.x*blockDim.x + threadIdx.x;
  if(i < NN){
    int c = cluster[i];
    #pragma unroll 4
    for(int f = 0; f < CATF; ++f)
      atomicAdd(&newx[(size_t)c*CATF + f], x[(size_t)i*CATF + f]);
  }
}
__global__ void k_newx_scale(float* __restrict__ newx, const float* __restrict__ csc){
  int idx = blockIdx.x*blockDim.x + threadIdx.x;
  if(idx < NN*CATF) newx[idx] *= csc[idx / CATF];
}
// Dedupe: key = (valid ? cs*N+cd : N*N) << 18 | e  (stable ascending sort)
__global__ void k_dedupe_key(const int* __restrict__ s, const int* __restrict__ d,
                             const float* __restrict__ w, const int* __restrict__ cluster,
                             int* __restrict__ ns0, int* __restrict__ nd0,
                             u64* __restrict__ keys){
  int e = blockIdx.x*blockDim.x + threadIdx.x;
  if(e < EE){
    int cs = cluster[s[e]], cd = cluster[d[e]];
    ns0[e] = cs; nd0[e] = cd;
    u64 kk = (w[e] > 0.f) ? ((u64)cs * (u64)NN + (u64)cd)
                          : ((u64)NN * (u64)NN);
    keys[e] = (kk << 18) | (unsigned)e;
  }
}
__global__ void k_dedupe_unpack(const u64* __restrict__ keys, const int* __restrict__ ns0,
                                const int* __restrict__ nd0, int* __restrict__ sN,
                                int* __restrict__ dN, float* __restrict__ wN){
  int i = blockIdx.x*blockDim.x + threadIdx.x;
  if(i < EE){
    u64 p = keys[i];
    int idx = (int)(p & 0x3FFFFu);
    u64 kk = p >> 18;
    bool uniq = (kk < (u64)NN * (u64)NN) &&
                (i == 0 || (keys[i-1] >> 18) != kk);
    sN[i] = ns0[idx];
    dN[i] = nd0[idx];
    wN[i] = uniq ? 1.f : 0.f;
  }
}

// ---------------------------------------------------------------------------
// Unpool + head
// ---------------------------------------------------------------------------
__global__ void k_unpool(const float* __restrict__ hin, const int* __restrict__ cl,
                         const float* __restrict__ csc, float* __restrict__ hout){
  int idx = blockIdx.x*blockDim.x + threadIdx.x;
  if(idx < NN*HIDN){
    int i = idx >> 6, f = idx & 63;
    int c = cl[i];
    hout[idx] = hin[(size_t)c*HIDN + f] / csc[c];
  }
}
__global__ void k_final(const float* __restrict__ xcat, const float* __restrict__ fw2,
                        const float* __restrict__ fb2, float* __restrict__ out){
  int i = blockIdx.x*blockDim.x + threadIdx.x;
  if(i < NN){
    float acc = fb2[0];
    const float* xr = xcat + (size_t)i*CATF;
    #pragma unroll 4
    for(int j = 0; j < CATF; ++j) acc += xr[j]*fw2[j];
    out[i] = 1.f / (1.f + expf(-acc));
  }
}

// ---------------------------------------------------------------------------
// Host-side orchestration helpers
// ---------------------------------------------------------------------------
static void sort_u64(u64* keys, hipStream_t st){
  for(unsigned k = 2; k <= EE; k <<= 1)
    for(unsigned j = k >> 1; j > 0; j >>= 1)
      k_bitonic<<<EE/256, 256, 0, st>>>(keys, j, k);
}

template<int K>
static void run_gcn(const float* X, const _Float16* Wh, const float* bias,
                    const int* s, const int* d, const float* w,
                    float* tmp64, float* deg, float* dis, float* out, hipStream_t st){
  k_wmma_mm<K><<<NN/16, 128, 0, st>>>(X, Wh, nullptr, tmp64, 0);
  k_fill_f32 <<<CDIV(NN,256), 256, 0, st>>>(deg, 1.f, NN);
  k_deg      <<<CDIV(EE,256), 256, 0, st>>>(s, d, w, deg);
  k_dis      <<<CDIV(NN,256), 256, 0, st>>>(deg, dis);
  k_self_init<<<CDIV(NN*HIDN,256), 256, 0, st>>>(tmp64, dis, out);
  k_edge_agg <<<CDIV(EE,256), 256, 0, st>>>(tmp64, s, d, w, dis, out);
  k_bias_relu<<<CDIV(NN*HIDN,256), 256, 0, st>>>(out, bias);
}

struct PoolScratch {
  float *r, *m, *den, *score, *scsrt;
  u64* keys;
  int *ssrt, *dsrt, *vsrt, *okA, *cidA, *assigned, *partial, *bsums, *kdev, *ns0, *nd0;
};

static void run_pool(const float* X, const int* s, const int* d, const float* w,
                     const float* pw, const float* pb,
                     float* newX, int* sN, int* dN, float* wN,
                     int* cluster, float* csc, const PoolScratch& t, hipStream_t st){
  k_pool_raw<<<CDIV(EE,256),256,0,st>>>(X, s, d, w, pw, pb, t.r);
  k_fill_f32<<<CDIV(NN,256),256,0,st>>>(t.m, -3.0e38f, NN);
  k_segmax  <<<CDIV(EE,256),256,0,st>>>(t.r, d, t.m);
  k_fill_f32<<<CDIV(NN,256),256,0,st>>>(t.den, 0.f, NN);
  k_segden  <<<CDIV(EE,256),256,0,st>>>(t.r, d, w, t.m, t.den);
  k_score   <<<CDIV(EE,256),256,0,st>>>(t.r, d, w, t.m, t.den, t.score);

  k_sortkey_score<<<CDIV(EE,256),256,0,st>>>(t.score, t.keys);
  sort_u64(t.keys, st);
  k_gather_sorted<<<CDIV(EE,256),256,0,st>>>(t.keys, s, d, w, t.score,
                                             t.ssrt, t.dsrt, t.vsrt, t.scsrt);
  k_fill_i32<<<CDIV(NN,256),256,0,st>>>(t.assigned, 0, NN);
  k_greedy  <<<1,1,0,st>>>(t.ssrt, t.dsrt, t.vsrt, t.assigned, t.okA, t.cidA, t.kdev);

  k_fill_i32<<<CDIV(NN,256),256,0,st>>>(cluster, 0, NN);
  k_fill_f32<<<CDIV(NN,256),256,0,st>>>(csc, 1.f, NN);
  k_scatter_cluster<<<CDIV(EE,256),256,0,st>>>(t.ssrt, t.dsrt, t.okA, t.cidA, t.scsrt,
                                               cluster, csc);
  k_scan_local   <<<NN/256, 256, 0, st>>>(t.assigned, t.partial, t.bsums);
  k_scan_block   <<<1, 256, 0, st>>>(t.bsums);
  k_final_cluster<<<CDIV(NN,256),256,0,st>>>(t.assigned, t.partial, t.bsums, t.kdev, cluster);

  k_fill_f32 <<<CDIV(NN*CATF,256),256,0,st>>>(newX, 0.f, NN*CATF);
  k_newx_sum <<<CDIV(NN,256),256,0,st>>>(X, cluster, newX);
  k_newx_scale<<<CDIV(NN*CATF,256),256,0,st>>>(newX, csc);

  k_dedupe_key<<<CDIV(EE,256),256,0,st>>>(s, d, w, cluster, t.ns0, t.nd0, t.keys);
  sort_u64(t.keys, st);
  k_dedupe_unpack<<<CDIV(EE,256),256,0,st>>>(t.keys, t.ns0, t.nd0, sN, dN, wN);
}

// ---------------------------------------------------------------------------
// Entry point
// ---------------------------------------------------------------------------
extern "C" void kernel_launch(void* const* d_in, const int* in_sizes, int n_in,
                              void* d_out, int out_size, void* d_ws, size_t ws_size,
                              hipStream_t stream){
  const float* x    = (const float*)d_in[0];
  const int*   edges= (const int*)  d_in[1];
  const float* cw1  = (const float*)d_in[2];  const float* cb1 = (const float*)d_in[3];
  const float* pw1  = (const float*)d_in[4];  const float* pb1 = (const float*)d_in[5];
  const float* pw2  = (const float*)d_in[6];  const float* pb2 = (const float*)d_in[7];
  const float* pw3  = (const float*)d_in[8];  const float* pb3 = (const float*)d_in[9];
  const float* cw2  = (const float*)d_in[10]; const float* cb2 = (const float*)d_in[11];
  const float* cw3  = (const float*)d_in[12]; const float* cb3 = (const float*)d_in[13];
  const float* cw4  = (const float*)d_in[14]; const float* cb4 = (const float*)d_in[15];
  const float* fw1  = (const float*)d_in[16]; const float* fb1 = (const float*)d_in[17];
  const float* fw2  = (const float*)d_in[18]; const float* fb2 = (const float*)d_in[19];
  float* out = (float*)d_out;

  char* p = (char*)d_ws;
  auto take = [&](size_t bytes)->char*{
    char* q = p; p += (bytes + 255) & ~(size_t)255; return q;
  };

  // f16 weights
  _Float16* whc1 = (_Float16*)take((size_t)FINF*64*2);
  _Float16* whc2 = (_Float16*)take((size_t)CATF*64*2);
  _Float16* whc3 = (_Float16*)take((size_t)CAT2*64*2);
  _Float16* whc4 = (_Float16*)take((size_t)CAT2*64*2);
  _Float16* whf1 = (_Float16*)take((size_t)CATF*64*2);
  // node feature buffers
  float* tmp64 = (float*)take((size_t)NN*64*4);
  float* h64   = (float*)take((size_t)NN*64*4);
  float* hupA  = (float*)take((size_t)NN*64*4);
  float* hupB  = (float*)take((size_t)NN*64*4);
  float* x0cat = (float*)take((size_t)NN*CATF*4);
  float* xa    = (float*)take((size_t)NN*CATF*4);
  float* xb    = (float*)take((size_t)NN*CATF*4);
  float* xc    = (float*)take((size_t)NN*CATF*4);
  float* c160  = (float*)take((size_t)NN*CAT2*4);
  // per-node scalars
  float* deg  = (float*)take((size_t)NN*4);
  float* dis  = (float*)take((size_t)NN*4);
  float* csc1 = (float*)take((size_t)NN*4);
  float* csc2 = (float*)take((size_t)NN*4);
  float* csc3 = (float*)take((size_t)NN*4);
  int*   cl1  = (int*)  take((size_t)NN*4);
  int*   cl2  = (int*)  take((size_t)NN*4);
  int*   cl3  = (int*)  take((size_t)NN*4);
  // edge lists per level
  int* s0=(int*)take((size_t)EE*4); int* d0=(int*)take((size_t)EE*4); float* w0=(float*)take((size_t)EE*4);
  int* s1=(int*)take((size_t)EE*4); int* d1=(int*)take((size_t)EE*4); float* w1=(float*)take((size_t)EE*4);
  int* s2=(int*)take((size_t)EE*4); int* d2=(int*)take((size_t)EE*4); float* w2=(float*)take((size_t)EE*4);
  int* s3=(int*)take((size_t)EE*4); int* d3=(int*)take((size_t)EE*4); float* w3=(float*)take((size_t)EE*4);
  // pool scratch
  PoolScratch t;
  t.r      = (float*)take((size_t)EE*4);
  t.m      = (float*)take((size_t)NN*4);
  t.den    = (float*)take((size_t)NN*4);
  t.score  = (float*)take((size_t)EE*4);
  t.scsrt  = (float*)take((size_t)EE*4);
  t.keys   = (u64*)  take((size_t)EE*8);
  t.ssrt   = (int*)  take((size_t)EE*4);
  t.dsrt   = (int*)  take((size_t)EE*4);
  t.vsrt   = (int*)  take((size_t)EE*4);
  t.okA    = (int*)  take((size_t)EE*4);
  t.cidA   = (int*)  take((size_t)EE*4);
  t.assigned=(int*)  take((size_t)NN*4);
  t.partial= (int*)  take((size_t)NN*4);
  t.bsums  = (int*)  take(256*4);
  t.kdev   = (int*)  take(256);
  t.ns0    = (int*)  take((size_t)EE*4);
  t.nd0    = (int*)  take((size_t)EE*4);

  hipStream_t st = stream;

  // Convert weights to f16 once per launch.
  k_cvt_f16<<<CDIV(FINF*64,256),256,0,st>>>(cw1, whc1, FINF*64);
  k_cvt_f16<<<CDIV(CATF*64,256),256,0,st>>>(cw2, whc2, CATF*64);
  k_cvt_f16<<<CDIV(CAT2*64,256),256,0,st>>>(cw3, whc3, CAT2*64);
  k_cvt_f16<<<CDIV(CAT2*64,256),256,0,st>>>(cw4, whc4, CAT2*64);
  k_cvt_f16<<<CDIV(CATF*64,256),256,0,st>>>(fw1, whf1, CATF*64);

  k_split_edges<<<CDIV(EE,256),256,0,st>>>(edges, s0, d0, w0);

  // GCN1: h = relu(gcn(x, cw1)); xcat = [x, h]
  run_gcn<FINF>(x, whc1, cb1, s0, d0, w0, tmp64, deg, dis, h64, st);
  k_concat2<<<CDIV(NN*CATF,256),256,0,st>>>(x, FINF, h64, HIDN, x0cat, NN);

  // Three EdgePool rounds.
  run_pool(x0cat, s0, d0, w0, pw1, pb1, xa, s1, d1, w1, cl1, csc1, t, st);
  run_pool(xa,    s1, d1, w1, pw2, pb2, xb, s2, d2, w2, cl2, csc2, t, st);
  run_pool(xb,    s2, d2, w2, pw3, pb3, xc, s3, d3, w3, cl3, csc3, t, st);

  // Coarse GCN stack (xp = xc).
  run_gcn<CATF>(xc, whc2, cb2, s3, d3, w3, tmp64, deg, dis, h64, st);
  k_concat2<<<CDIV(NN*CAT2,256),256,0,st>>>(xc, CATF, h64, HIDN, c160, NN);
  run_gcn<CAT2>(c160, whc3, cb3, s3, d3, w3, tmp64, deg, dis, h64, st);
  k_concat2<<<CDIV(NN*CAT2,256),256,0,st>>>(xc, CATF, h64, HIDN, c160, NN);
  run_gcn<CAT2>(c160, whc4, cb4, s3, d3, w3, tmp64, deg, dis, h64, st);

  // Unpool: (h/sc3)[cl3] -> (/sc2)[cl2] -> (/sc1)[cl1]
  k_unpool<<<CDIV(NN*HIDN,256),256,0,st>>>(h64,  cl3, csc3, hupA);
  k_unpool<<<CDIV(NN*HIDN,256),256,0,st>>>(hupA, cl2, csc2, hupB);
  k_unpool<<<CDIV(NN*HIDN,256),256,0,st>>>(hupB, cl1, csc1, hupA);

  // Head: relu([x,h] @ fw1 + fb1); sigmoid([x,h'] @ fw2 + fb2)
  k_concat2<<<CDIV(NN*CATF,256),256,0,st>>>(x, FINF, hupA, HIDN, xa, NN);
  k_wmma_mm<CATF><<<NN/16,128,0,st>>>(xa, whf1, fb1, h64, 1);
  k_concat2<<<CDIV(NN*CATF,256),256,0,st>>>(x, FINF, h64, HIDN, xb, NN);
  k_final<<<CDIV(NN,256),256,0,st>>>(xb, fw2, fb2, out);

  (void)in_sizes; (void)n_in; (void)out_size; (void)ws_size;
}
